// EB_33466385170816
// MI455X (gfx1250) — matrix operand order repaired and checked by reading, not compiled, not run on hardware
//
#include <hip/hip_runtime.h>
#include <hip/hip_bf16.h>

// ---------------------------------------------------------------------------
// CDNA5 (gfx1250, wave32). All GEMMs -> v_wmma_f32_16x16x32_bf16.
// Block tile 128(M) x 64(N), BK=32, 256 threads = 8 waves, wave tile 32x32.
// Register-staged double buffering; two-phase epilogues (batched loads).
// Weights pre-transposed to bf16 [N][K] with LN gamma/beta folded in.
// NGUARD template param: N-bounds checks only instantiated where N%64 != 0.
// REQUIREMENT: M % 128 == 0, K % 32 == 0, N % 4 == 0 (all hold here).
// ---------------------------------------------------------------------------

typedef __attribute__((ext_vector_type(16))) __bf16 v16bf;
typedef __attribute__((ext_vector_type(8)))  __bf16 v8bf;
typedef __attribute__((ext_vector_type(4)))  __bf16 v4bf;
typedef __attribute__((ext_vector_type(2)))  __bf16 v2bf;
typedef __attribute__((ext_vector_type(8)))  float  v8f;

struct __align__(16) F4 { float v[4]; };

#define DIMD   1024
#define NHW    49
#define INNERD 512
#define KMAPN  392   // HEADS * 49

__device__ __forceinline__ float sigmoidf_(float v) {
    return 1.0f / (1.0f + __expf(-v));
}

// ------------------------------- A loaders ---------------------------------
// fetch4() issues global loads; cvt4() converts later (after the WMMAs).
struct ALoadF32 {
    const float* A; int lda;
    using Raw = F4;
    __device__ Raw fetch4(int r, int k) const { return *(const F4*)(A + r * lda + k); }
    __device__ static void cvt4(const Raw& x, __bf16* o) {
#pragma unroll
        for (int i = 0; i < 4; ++i) o[i] = (__bf16)x.v[i];
    }
};
struct ALoadBF16 {
    const __bf16* A; int lda;
    using Raw = v4bf;
    __device__ Raw fetch4(int r, int k) const { return *(const v4bf*)(A + r * lda + k); }
    __device__ static void cvt4(const Raw& x, __bf16* o) {
#pragma unroll
        for (int i = 0; i < 4; ++i) o[i] = x[i];
    }
};
struct ALoadQK {           // concat(q - oq, oq); halves are 32-aligned in K
    const float* q; const float* oq;
    struct Raw { F4 a, b; int sub; };
    __device__ Raw fetch4(int r, int k) const {
        Raw rw;
        if (k < DIMD) {
            rw.a = *(const F4*)(q  + r * DIMD + k);
            rw.b = *(const F4*)(oq + r * DIMD + k);
            rw.sub = 1;
        } else {
            rw.a = *(const F4*)(oq + r * DIMD + (k - DIMD));
            rw.sub = 0;
        }
        return rw;
    }
    __device__ static void cvt4(const Raw& x, __bf16* o) {
#pragma unroll
        for (int i = 0; i < 4; ++i)
            o[i] = (__bf16)(x.sub ? (x.a.v[i] - x.b.v[i]) : x.a.v[i]);
    }
};
struct ALoadLNS {          // (x - mean) * rstd ; gamma/beta folded into weights
    const float* X; const float* stats; int lda;
    struct Raw { F4 x; float m, rs; };
    __device__ Raw fetch4(int r, int k) const {
        Raw rw;
        rw.x  = *(const F4*)(X + r * lda + k);
        rw.m  = stats[2 * r];
        rw.rs = stats[2 * r + 1];
        return rw;
    }
    __device__ static void cvt4(const Raw& x, __bf16* o) {
#pragma unroll
        for (int i = 0; i < 4; ++i) o[i] = (__bf16)((x.x.v[i] - x.m) * x.rs);
    }
};

// ------------------------------- Epilogues ---------------------------------
// Two-phase: pre(r,c) performs all global reads (batched by the driver so the
// loads pipeline), post(r,c,acc,pre) does math + store.
struct EpiBiasF32 {
    float* C; int ldc; const float* bias;
    using Pre = float;
    __device__ Pre pre(int r, int c) const { return bias[c]; }
    __device__ void post(int r, int c, float acc, Pre p) const {
        C[r * ldc + c] = acc + p;
    }
};
struct EpiSiluBF16 {
    __bf16* C; int ldc; const float* bias;
    using Pre = float;
    __device__ Pre pre(int r, int c) const { return bias[c]; }
    __device__ void post(int r, int c, float acc, Pre p) const {
        float v = acc + p;
        C[r * ldc + c] = (__bf16)(v * sigmoidf_(v));
    }
};
struct EpiUpdate {         // h2 -> new_q (ws) | new_k (out)
    float* newq; float* newk; const float* val; const float* q; const float* oq;
    const float* bias;
    struct Pre { float v, rq, bi; };
    __device__ Pre pre(int r, int c) const {
        Pre p;
        p.bi = bias[c];
        if (c < DIMD) {
            p.v  = val[r * DIMD + c];
            p.rq = q[r * DIMD + c];
        } else {
            int c2 = c - DIMD;
            p.v  = val[r * DIMD + c2];
            p.rq = oq[r * DIMD + c2];
        }
        return p;
    }
    __device__ void post(int r, int c, float acc, Pre p) const {
        float s = sigmoidf_(acc + p.bi);
        float o = s * p.v + p.rq;
        if (c < DIMD) newq[r * DIMD + c] = o;
        else          newk[r * DIMD + (c - DIMD)] = o;
    }
};
struct EpiSigmoid {        // s = sigmoid(acc + folded bias), ldc = 392
    float* S; int ldc; const float* bias;
    using Pre = float;
    __device__ Pre pre(int r, int c) const { return bias[c]; }
    __device__ void post(int r, int c, float acc, Pre p) const {
        S[r * ldc + c] = sigmoidf_(acc + p);
    }
};
struct EpiGateValue {      // value = (acc + bias') * sig(kmap[b, head, p]) -> bf16
    __bf16* V; const float* S; const float* bias;
    struct Pre { float gate, bi; };
    __device__ Pre pre(int r, int c) const {
        Pre p;
        int b    = r / NHW;
        int sp   = r - b * NHW;
        int head = c >> 6;
        p.gate = S[b * KMAPN + head * NHW + sp];
        p.bi   = bias[c];
        return p;
    }
    __device__ void post(int r, int c, float acc, Pre p) const {
        V[r * INNERD + c] = (__bf16)((acc + p.bi) * p.gate);
    }
};
struct EpiAddRes {
    float* C; const float* R; int ldc;
    using Pre = float;
    __device__ Pre pre(int r, int c) const { return R[r * ldc + c]; }
    __device__ void post(int r, int c, float acc, Pre p) const {
        C[r * ldc + c] = acc + p;
    }
};

// ------------------------- WMMA GEMM (bf16 -> f32) -------------------------
template <class AL, class EP, bool NGUARD>
__launch_bounds__(256)
__global__ void gemm_wmma_k(AL al, const __bf16* __restrict__ Bt, int ldb, EP ep,
                            int M, int N, int K) {
    constexpr int BM = 128, BN = 64, BK = 32;
    constexpr int AS = 40;   // 80B pitch -> conflict-free 16B fragment reads
    constexpr int BS = 40;

    __shared__ __align__(16) __bf16 ldsA[2][BM * AS];
    __shared__ __align__(16) __bf16 ldsB[2][BN * BS];

    const int tid   = threadIdx.x;
    const int lane  = tid & 31;
    const int wv    = tid >> 5;
    const int wrow  = (wv >> 1) * 32;
    const int wcol  = (wv & 1) * 32;
    const int lhalf = lane >> 4;
    const int lidx  = lane & 15;

    const int m0 = blockIdx.y * BM;
    const int n0 = blockIdx.x * BN;

    const int ar  = (tid >> 3);          // A row base (+32 per chunk)
    const int ac  = (tid & 7) * 4;       // A col quad
    const int bn_ = (tid >> 3);          // B n base (+32 for 2nd chunk)
    const int bk4 = (tid & 7) * 4;       // B k quad

    typename AL::Raw araw[4];
    v4bf braw[2];

    auto fetch = [&](int k0) {
#pragma unroll
        for (int i = 0; i < 4; ++i)
            araw[i] = al.fetch4(m0 + ar + i * 32, k0 + ac);
#pragma unroll
        for (int j = 0; j < 2; ++j) {
            int n = bn_ + j * 32;
            if (!NGUARD || (n0 + n < N)) {
                braw[j] = *(const v4bf*)(Bt + (n0 + n) * ldb + k0 + bk4);
            } else {
                v4bf z = { (__bf16)0.f, (__bf16)0.f, (__bf16)0.f, (__bf16)0.f };
                braw[j] = z;
            }
        }
    };
    auto commit = [&](int buf) {
#pragma unroll
        for (int i = 0; i < 4; ++i) {
            __bf16 o[4];
            AL::cvt4(araw[i], o);
            v4bf v = { o[0], o[1], o[2], o[3] };
            *(v4bf*)&ldsA[buf][(ar + i * 32) * AS + ac] = v;
        }
#pragma unroll
        for (int j = 0; j < 2; ++j)
            *(v4bf*)&ldsB[buf][(bn_ + j * 32) * BS + bk4] = braw[j];
    };

    v8f acc[2][2];
#pragma unroll
    for (int mt = 0; mt < 2; ++mt)
#pragma unroll
        for (int nt = 0; nt < 2; ++nt)
#pragma unroll
            for (int j = 0; j < 8; ++j) acc[mt][nt][j] = 0.0f;

    fetch(0);
    commit(0);
    __syncthreads();

    const int nsteps = K / BK;
    for (int s = 0; s < nsteps; ++s) {
        const int cur = s & 1;
        const bool has = (s + 1 < nsteps);
        if (has) fetch((s + 1) * BK);          // global loads in flight over WMMAs

        v16bf afrag[2], bfrag[2];
#pragma unroll
        for (int mt = 0; mt < 2; ++mt) {
            const __bf16* ap = &ldsA[cur][(wrow + mt * 16 + lidx) * AS];
            v8bf lo = *(const v8bf*)(ap + lhalf * 8);
            v8bf hi = *(const v8bf*)(ap + 16 + lhalf * 8);
#pragma unroll
            for (int i = 0; i < 8; ++i) { afrag[mt][i] = lo[i]; afrag[mt][8 + i] = hi[i]; }
        }
#pragma unroll
        for (int nt = 0; nt < 2; ++nt) {
            const __bf16* bp = &ldsB[cur][(wcol + nt * 16 + lidx) * BS];
            v8bf lo = *(const v8bf*)(bp + lhalf * 8);
            v8bf hi = *(const v8bf*)(bp + 16 + lhalf * 8);
#pragma unroll
            for (int i = 0; i < 8; ++i) { bfrag[nt][i] = lo[i]; bfrag[nt][8 + i] = hi[i]; }
        }
#pragma unroll
        for (int mt = 0; mt < 2; ++mt)
#pragma unroll
            for (int nt = 0; nt < 2; ++nt)
                acc[mt][nt] = __builtin_amdgcn_wmma_f32_16x16x32_bf16(
                    false, afrag[mt], false, bfrag[nt],
                    (short)0, acc[mt][nt], false, false);

        if (has) commit(cur ^ 1);              // wait loads, cvt, LDS stores
        __syncthreads();
    }

    // Two-phase epilogue, batched per m-tile: 16 loads in flight, one wait.
#pragma unroll
    for (int mt = 0; mt < 2; ++mt) {
        typename EP::Pre pres[2][8];
#pragma unroll
        for (int nt = 0; nt < 2; ++nt)
#pragma unroll
            for (int j = 0; j < 8; ++j) {
                int r = m0 + wrow + mt * 16 + lhalf * 8 + j;
                int c = n0 + wcol + nt * 16 + lidx;
                if (!NGUARD || c < N) pres[nt][j] = ep.pre(r, c);
            }
#pragma unroll
        for (int nt = 0; nt < 2; ++nt)
#pragma unroll
            for (int j = 0; j < 8; ++j) {
                int r = m0 + wrow + mt * 16 + lhalf * 8 + j;
                int c = n0 + wcol + nt * 16 + lidx;
                if (!NGUARD || c < N) ep.post(r, c, acc[mt][nt][j], pres[nt][j]);
            }
    }
}

// ------------------- weight pre-transpose / LN fold (bf16) -----------------
// Wt[n*K + k] = bf16( W[k*N + n] * (g ? g[k] : 1) )
__global__ void wt_cvt_k(const float* __restrict__ W, const float* __restrict__ g,
                         __bf16* __restrict__ Wt, int K, int N) {
    int idx = blockIdx.x * blockDim.x + threadIdx.x;
    if (idx >= N * K) return;
    int n = idx / K, k = idx - n * K;
    float v = W[k * N + n];
    if (g) v *= g[k];
    Wt[idx] = (__bf16)v;
}

// out[n] = bias[n] + sum_k beta[k] * W[k*N + n]   (one wave per n)
__global__ void bias_fold_k(const float* __restrict__ W, const float* __restrict__ beta,
                            const float* __restrict__ bias, float* __restrict__ out,
                            int K, int N) {
    int n    = blockIdx.x * (blockDim.x >> 5) + (threadIdx.x >> 5);
    int lane = threadIdx.x & 31;
    if (n >= N) return;
    float s = 0.0f;
    for (int k = lane; k < K; k += 32) s += beta[k] * W[k * N + n];
#pragma unroll
    for (int off = 16; off > 0; off >>= 1) s += __shfl_xor(s, off, 32);
    if (lane == 0) out[n] = bias[n] + s;
}

// --------------------- LayerNorm row stats (mean, rstd) --------------------
__global__ void ln_stats_k(const float* __restrict__ X, float* __restrict__ stats,
                           int nrows, int D) {
    int row  = blockIdx.x * (blockDim.x >> 5) + (threadIdx.x >> 5);
    int lane = threadIdx.x & 31;
    if (row >= nrows) return;
    const float* x = X + row * D;
    float s = 0.0f, s2 = 0.0f;
    for (int i = lane; i < D; i += 32) { float v = x[i]; s += v; s2 += v * v; }
#pragma unroll
    for (int off = 16; off > 0; off >>= 1) {
        s  += __shfl_xor(s,  off, 32);
        s2 += __shfl_xor(s2, off, 32);
    }
    if (lane == 0) {
        float m   = s / D;
        float var = s2 / D - m * m;
        stats[2 * row]     = m;
        stats[2 * row + 1] = rsqrtf(var + 1e-5f);
    }
}

// ---------------- mean over 49 spatial positions (paired bf16) -------------
__global__ void mean49_k(const __bf16* __restrict__ V, __bf16* __restrict__ Vm) {
    int idx = blockIdx.x * blockDim.x + threadIdx.x;   // over B * 256 n-pairs
    if (idx >= 1024 * (INNERD / 2)) return;
    int b = idx >> 8, np = (idx & 255) * 2;
    const __bf16* p = V + b * NHW * INNERD + np;
    float s0 = 0.0f, s1 = 0.0f;
#pragma unroll
    for (int i = 0; i < NHW; ++i) {
        v2bf t = *(const v2bf*)(p + i * INNERD);
        s0 += (float)t[0];
        s1 += (float)t[1];
    }
    v2bf o = { (__bf16)(s0 * (1.0f / 49.0f)), (__bf16)(s1 * (1.0f / 49.0f)) };
    *(v2bf*)(Vm + b * INNERD + np) = o;
}

// ---------------------------------------------------------------------------
extern "C" void kernel_launch(void* const* d_in, const int* in_sizes, int n_in,
                              void* d_out, int out_size, void* d_ws, size_t ws_size,
                              hipStream_t stream) {
    const int B  = 1024;
    const int MX = B * NHW;            // 50176

    const float* x     = (const float*)d_in[0];
    const float* q     = (const float*)d_in[1];
    const float* oq    = (const float*)d_in[2];
    const float* oq_v  = (const float*)d_in[3];
    const float* Wq1   = (const float*)d_in[4];
    const float* bq1   = (const float*)d_in[5];
    const float* Wq2   = (const float*)d_in[6];
    const float* bq2   = (const float*)d_in[7];
    const float* Wv    = (const float*)d_in[8];
    const float* bv    = (const float*)d_in[9];
    const float* gx    = (const float*)d_in[10];
    const float* bx_ln = (const float*)d_in[11];
    const float* Wx    = (const float*)d_in[12];
    const float* bx    = (const float*)d_in[13];
    const float* gk    = (const float*)d_in[14];
    const float* bk_ln = (const float*)d_in[15];
    const float* Wk    = (const float*)d_in[16];
    const float* bk    = (const float*)d_in[17];
    const float* Wtv   = (const float*)d_in[18];
    const float* Wtn   = (const float*)d_in[19];

    float* out   = (float*)d_out;
    float* x_out = out;                              // [B*49, 1024]
    float* q_out = out + (size_t)MX * DIMD;          // [B, 1024]
    float* new_k = q_out + (size_t)B * DIMD;         // [B, 1024]

    // ---- workspace carve-out ----
    size_t off = 0;
    auto carve = [&](size_t bytes) {
        size_t o = off; off += (bytes + 255) & ~(size_t)255; return o;
    };
    char* ws = (char*)d_ws;
    float*  val_ws   = (float*) (ws + carve((size_t)B * DIMD * 4));
    float*  newq_ws  = (float*) (ws + carve((size_t)B * DIMD * 4));
    __bf16* h1_ws    = (__bf16*)(ws + carve((size_t)B * 256 * 2));
    float*  s_ws     = (float*) (ws + carve((size_t)B * KMAPN * 4));
    float*  statsX   = (float*) (ws + carve((size_t)MX * 2 * 4));
    float*  statsQ   = (float*) (ws + carve((size_t)B * 2 * 4));
    __bf16* vmean_ws = (__bf16*)(ws + carve((size_t)B * INNERD * 2));
    __bf16* WvT   = (__bf16*)(ws + carve((size_t)DIMD * DIMD * 2));
    __bf16* Wq1T  = (__bf16*)(ws + carve((size_t)256 * 2 * DIMD * 2));
    __bf16* Wq2T  = (__bf16*)(ws + carve((size_t)2 * DIMD * 256 * 2));
    __bf16* WkT   = (__bf16*)(ws + carve((size_t)KMAPN * DIMD * 2));
    __bf16* WxT   = (__bf16*)(ws + carve((size_t)INNERD * DIMD * 2));
    __bf16* WtvT  = (__bf16*)(ws + carve((size_t)DIMD * INNERD * 2));
    __bf16* WtnT  = (__bf16*)(ws + carve((size_t)DIMD * INNERD * 2));
    float*  bkF   = (float*) (ws + carve((size_t)KMAPN * 4));
    float*  bxF   = (float*) (ws + carve((size_t)INNERD * 4));
    __bf16* value_ws = (__bf16*)(ws + carve((size_t)MX * INNERD * 2));
    (void)ws_size; (void)in_sizes; (void)n_in; (void)out_size;

    auto grid    = [](int M, int N) { return dim3((N + 63) / 64, (M + 127) / 128); };
    auto cdiv256 = [](int n) { return (n + 255) / 256; };

    // ---- weight preprocessing (transpose -> bf16, fold LN gamma/beta) ----
    wt_cvt_k<<<cdiv256(DIMD * DIMD),    256, 0, stream>>>(Wv,  nullptr, WvT,  DIMD,     DIMD);
    wt_cvt_k<<<cdiv256(2 * DIMD * 256), 256, 0, stream>>>(Wq1, nullptr, Wq1T, 2 * DIMD, 256);
    wt_cvt_k<<<cdiv256(256 * 2 * DIMD), 256, 0, stream>>>(Wq2, nullptr, Wq2T, 256,      2 * DIMD);
    wt_cvt_k<<<cdiv256(DIMD * KMAPN),   256, 0, stream>>>(Wk,  gk,      WkT,  DIMD,     KMAPN);
    wt_cvt_k<<<cdiv256(DIMD * INNERD),  256, 0, stream>>>(Wx,  gx,      WxT,  DIMD,     INNERD);
    wt_cvt_k<<<cdiv256(INNERD * DIMD),  256, 0, stream>>>(Wtv, nullptr, WtvT, INNERD,   DIMD);
    wt_cvt_k<<<cdiv256(INNERD * DIMD),  256, 0, stream>>>(Wtn, nullptr, WtnT, INNERD,   DIMD);
    bias_fold_k<<<(KMAPN + 7) / 8,  256, 0, stream>>>(Wk, bk_ln, bk, bkF, DIMD, KMAPN);
    bias_fold_k<<<(INNERD + 7) / 8, 256, 0, stream>>>(Wx, bx_ln, bx, bxF, DIMD, INNERD);

    // [0] LN stats for x
    ln_stats_k<<<(MX + 7) / 8, 256, 0, stream>>>(x, statsX, MX, DIMD);

    // [1] val = oq_v @ Wv + bv
    gemm_wmma_k<ALoadF32, EpiBiasF32, false><<<grid(B, DIMD), 256, 0, stream>>>(
        ALoadF32{oq_v, DIMD}, WvT, DIMD,
        EpiBiasF32{val_ws, DIMD, bv}, B, DIMD, DIMD);

    // [2] h1 = silu(concat(q-oq, oq) @ Wq1 + bq1)   M=1024 N=256 K=2048
    gemm_wmma_k<ALoadQK, EpiSiluBF16, false><<<grid(B, 256), 256, 0, stream>>>(
        ALoadQK{q, oq}, Wq1T, 2 * DIMD,
        EpiSiluBF16{h1_ws, 256, bq1}, B, 256, 2 * DIMD);

    // [3] h = h1 @ Wq2 + bq2 -> new_q / new_k       M=1024 N=2048 K=256
    gemm_wmma_k<ALoadBF16, EpiUpdate, false><<<grid(B, 2 * DIMD), 256, 0, stream>>>(
        ALoadBF16{h1_ws, 256}, Wq2T, 256,
        EpiUpdate{newq_ws, new_k, val_ws, q, oq, bq2}, B, 2 * DIMD, 256);

    // [4] LN stats for new_q
    ln_stats_k<<<(B + 7) / 8, 256, 0, stream>>>(newq_ws, statsQ, B, DIMD);

    // [5] s = sigmoid(LN(new_q) @ Wk' + bk')        M=1024 N=392 K=1024 (guarded)
    gemm_wmma_k<ALoadLNS, EpiSigmoid, true><<<grid(B, KMAPN), 256, 0, stream>>>(
        ALoadLNS{newq_ws, statsQ, DIMD}, WkT, DIMD,
        EpiSigmoid{s_ws, KMAPN, bkF}, B, KMAPN, DIMD);

    // [6] value = (LN(x) @ Wx' + bx') * gate        M=50176 N=512 K=1024
    gemm_wmma_k<ALoadLNS, EpiGateValue, false><<<grid(MX, INNERD), 256, 0, stream>>>(
        ALoadLNS{x, statsX, DIMD}, WxT, DIMD,
        EpiGateValue{value_ws, s_ws, bxF}, MX, INNERD, DIMD);

    // [7] x_out = value @ Wtv + x                   M=50176 N=1024 K=512
    gemm_wmma_k<ALoadBF16, EpiAddRes, false><<<grid(MX, DIMD), 256, 0, stream>>>(
        ALoadBF16{value_ws, INNERD}, WtvT, INNERD,
        EpiAddRes{x_out, x, DIMD}, MX, DIMD, INNERD);

    // [8] vmean = mean_p value
    mean49_k<<<cdiv256(B * (INNERD / 2)), 256, 0, stream>>>(value_ws, vmean_ws);

    // [9] q_out = vmean @ Wtn + q                   M=1024 N=1024 K=512
    gemm_wmma_k<ALoadBF16, EpiAddRes, false><<<grid(B, DIMD), 256, 0, stream>>>(
        ALoadBF16{vmean_ws, INNERD}, WtnT, INNERD,
        EpiAddRes{q_out, q, DIMD}, B, DIMD, INNERD);
}
